// Multihead_Latent_Attention_70454643523650
// MI455X (gfx1250) — matrix operand order repaired and checked by reading, not compiled, not run on hardware
//
#include <hip/hip_runtime.h>

typedef __attribute__((ext_vector_type(16))) __bf16 v16bf;
typedef __attribute__((ext_vector_type(8)))  float  v8f;

#define D_MODEL 3072
#define NH      16
#define HD      192
#define DC      1024
#define BATCH   2
#define SEQ     2048
#define BT      4096   /* BATCH*SEQ */
#define LN_EPS  1e-5f

union FragU { v16bf v; uint4 q[2]; };

__device__ __forceinline__ unsigned short f2bf(float f) {
  unsigned u = __builtin_bit_cast(unsigned, f);
  u += 0x7FFFu + ((u >> 16) & 1u);
  return (unsigned short)(u >> 16);
}

__device__ __forceinline__ v8f zero8() {
  v8f z;
#pragma unroll
  for (int i = 0; i < 8; ++i) z[i] = 0.0f;
  return z;
}

// A-fragment (16x32, bf16). Arow already includes the per-lane row (lane&15).
// Layout: lanes 0-15 hold K = kk+0..7 and kk+16..23; lanes 16-31 hold +8.
__device__ __forceinline__ v16bf load_frag_a(const unsigned short* Arow, int kk, int lane) {
  int kb = kk + ((lane >> 4) << 3);
  const uint4* p = reinterpret_cast<const uint4*>(Arow + kb);
  FragU f; f.q[0] = p[0]; f.q[1] = p[2];   // elements kb..kb+7, kb+16..kb+23
  return f.v;
}

// B-fragment (32x16, bf16), NT form: Brow = &B[n*ldb] (n already includes lane&15).
// Layout: lanes 0-15 hold K = kk+0..7 and kk+16..23; lanes 16-31 hold +8.
__device__ __forceinline__ v16bf load_frag_b(const unsigned short* Brow, int kk, int lane) {
  int kb = kk + ((lane >> 4) << 3);
  const uint4* p = reinterpret_cast<const uint4*>(Brow + kb);
  FragU f; f.q[0] = p[0]; f.q[1] = p[2];   // elements kb..kb+7, kb+16..kb+23
  return f.v;
}

#define WMMA_BF16(a, b, c) \
  __builtin_amdgcn_wmma_f32_16x16x32_bf16(false, (a), false, (b), (short)0, (c), false, false)

/* ---------------- fp32 -> bf16 conversion kernels ---------------- */
__global__ __launch_bounds__(256) void cvt_bf16(const float* __restrict__ src,
                                                unsigned short* __restrict__ dst, size_t n) {
  for (size_t i = (size_t)blockIdx.x * blockDim.x + threadIdx.x; i < n;
       i += (size_t)gridDim.x * blockDim.x)
    dst[i] = f2bf(src[i]);
}

// dst[c*R + r] = bf16(src[r*C + c])   (transpose-convert, R x C -> C x R)
__global__ __launch_bounds__(256) void cvt_bf16_t(const float* __restrict__ src,
                                                  unsigned short* __restrict__ dst,
                                                  int R, int C) {
  size_t n = (size_t)R * C;
  for (size_t i = (size_t)blockIdx.x * blockDim.x + threadIdx.x; i < n;
       i += (size_t)gridDim.x * blockDim.x) {
    int r = (int)(i / C), c = (int)(i % C);
    dst[(size_t)c * R + r] = f2bf(src[i]);
  }
}

/* ---------------- NT GEMM: C[M,N] = A[M,K] * B[N,K]^T (+bias) ----------------
   block = 256 threads = 8 waves arranged 2(M) x 4(N); wave tile 32x64.
   Block tile 64 x 256. Requires M%64==0, N%256==0, K%32==0.                  */
template <bool OUT_BF16, bool HAS_BIAS>
__global__ __launch_bounds__(256)
void gemm_nt(const unsigned short* __restrict__ A, const unsigned short* __restrict__ Bm,
             void* __restrict__ Cv, const float* __restrict__ bias,
             int M, int N, int K, int lda, int ldb, int ldc) {
  const int lane = threadIdx.x & 31;
  const int wid  = threadIdx.x >> 5;
  const int m0 = blockIdx.y * 64  + (wid >> 2) * 32;
  const int n0 = blockIdx.x * 256 + (wid & 3) * 64;

  v8f acc[2][4];
#pragma unroll
  for (int mi = 0; mi < 2; ++mi)
#pragma unroll
    for (int ni = 0; ni < 4; ++ni) acc[mi][ni] = zero8();

  const unsigned short* Ar0 = A + (size_t)(m0 + (lane & 15)) * lda;
  const unsigned short* Ar1 = Ar0 + (size_t)16 * lda;
  const unsigned short* Br[4];
#pragma unroll
  for (int ni = 0; ni < 4; ++ni)
    Br[ni] = Bm + (size_t)(n0 + ni * 16 + (lane & 15)) * ldb;

  for (int kk = 0; kk < K; kk += 32) {
    v16bf a0 = load_frag_a(Ar0, kk, lane);
    v16bf a1 = load_frag_a(Ar1, kk, lane);
#pragma unroll
    for (int ni = 0; ni < 4; ++ni) {
      v16bf b = load_frag_b(Br[ni], kk, lane);
      acc[0][ni] = WMMA_BF16(a0, b, acc[0][ni]);
      acc[1][ni] = WMMA_BF16(a1, b, acc[1][ni]);
    }
  }

  const int cL   = lane & 15;
  const int rOff = (lane >> 4) << 3;
#pragma unroll
  for (int mi = 0; mi < 2; ++mi)
#pragma unroll
    for (int ni = 0; ni < 4; ++ni) {
      int col = n0 + ni * 16 + cL;
      float bv = HAS_BIAS ? bias[col] : 0.0f;
#pragma unroll
      for (int j = 0; j < 8; ++j) {
        int row = m0 + mi * 16 + rOff + j;
        float v = acc[mi][ni][j] + bv;
        if (OUT_BF16) ((unsigned short*)Cv)[(size_t)row * ldc + col] = f2bf(v);
        else          ((float*)Cv)[(size_t)row * ldc + col] = v;
      }
    }
}

/* ---------------- LayerNorm over rows of c (BT x DC) ---------------- */
__global__ __launch_bounds__(256)
void ln_kernel(const float* __restrict__ c, const float* __restrict__ g,
               const float* __restrict__ b, float* __restrict__ out_f32,
               unsigned short* __restrict__ out_bf16) {
  const int row = blockIdx.x;
  const float* cr = c + (size_t)row * DC;
  float vals[4], s = 0.f, ss = 0.f;
#pragma unroll
  for (int i = 0; i < 4; ++i) {
    float v = cr[threadIdx.x + i * 256];
    vals[i] = v; s += v; ss += v * v;
  }
#pragma unroll
  for (int off = 16; off >= 1; off >>= 1) {
    s  += __shfl_xor(s,  off, 32);
    ss += __shfl_xor(ss, off, 32);
  }
  __shared__ float red[16];
  const int lane = threadIdx.x & 31, wid = threadIdx.x >> 5;
  if (lane == 0) { red[wid] = s; red[8 + wid] = ss; }
  __syncthreads();
  float ts = 0.f, tss = 0.f;
#pragma unroll
  for (int w = 0; w < 8; ++w) { ts += red[w]; tss += red[8 + w]; }
  float mu  = ts * (1.0f / DC);
  float var = tss * (1.0f / DC) - mu * mu;
  float inv = rsqrtf(var + LN_EPS);
#pragma unroll
  for (int i = 0; i < 4; ++i) {
    int col = threadIdx.x + i * 256;
    float y = (vals[i] - mu) * inv * g[col] + b[col];
    out_f32[(size_t)row * DC + col]  = y;
    out_bf16[(size_t)row * DC + col] = f2bf(y);
  }
}

/* ---------------- Flash attention (causal) ----------------
   grid = (SEQ/128, BATCH*NH), block = 256 (8 waves).
   Each wave owns 16 q rows; K tiles staged via async global->LDS (ASYNCcnt),
   V tiles transposed into LDS manually.                                     */
__global__ __launch_bounds__(256)
void mla_attn(const unsigned short* __restrict__ xq,    // (BT, 3072) bf16 (queries = x)
              const unsigned short* __restrict__ keff,  // (BT, 3072) bf16 decompressed keys
              const unsigned short* __restrict__ vfull, // (BT, 3072) bf16 decompressed values
              unsigned short* __restrict__ ctx) {       // (BT, 3072) bf16 context out
  __shared__ __align__(16) unsigned short Kl[32 * 200];   // K tile  [s][d], pad 200
  __shared__ __align__(16) unsigned short Vt[192 * 40];   // V tile transposed [d][s], pad 40
  __shared__ __align__(16) unsigned short Pb[8 * 16 * 40];// per-wave P [row][col], pad 40

  const int lane = threadIdx.x & 31;
  const int wid  = threadIdx.x >> 5;
  const int tblk = blockIdx.x;
  const int b    = blockIdx.y >> 4;
  const int h    = blockIdx.y & 15;
  const int tbase = tblk * 128;
  const int twb   = tbase + wid * 16;          // this wave's first q row
  const size_t rowB = (size_t)b * SEQ;
  const int colh = h * HD;
  unsigned short* Pw = Pb + wid * 16 * 40;

  // Preload 6 Q fragments (16 x 192)
  v16bf qf[6];
  {
    const unsigned short* qrow =
        xq + (rowB + twb + (lane & 15)) * (size_t)D_MODEL + colh;
#pragma unroll
    for (int kc = 0; kc < 6; ++kc) qf[kc] = load_frag_a(qrow, kc * 32, lane);
  }

  v8f ctxa[12];
#pragma unroll
  for (int ti = 0; ti < 12; ++ti) ctxa[ti] = zero8();
  float mrow[8], lrow[8];
#pragma unroll
  for (int j = 0; j < 8; ++j) { mrow[j] = -1e30f; lrow[j] = 0.f; }

  const float scale = 0.07216878364870323f;  // 1/sqrt(192)
  const int cL   = lane & 15;
  const int rOff = (lane >> 4) << 3;
  const int smax = tbase + 127;

  for (int s0 = 0; s0 <= smax; s0 += 32) {
    __syncthreads();
    // Cooperative stage: K rows via async global->LDS DMA, V transposed manually
#pragma unroll
    for (int kload = 0; kload < 3; ++kload) {
      int idx = threadIdx.x + kload * 256;     // 0..767
      int s = idx / 24;
      int d = (idx % 24) * 8;
      size_t grow = (rowB + s0 + s) * (size_t)D_MODEL + colh + d;
      unsigned kldsoff = (unsigned)(size_t)(&Kl[s * 200 + d]);
      asm volatile("global_load_async_to_lds_b128 %0, %1, off"
                   :: "v"(kldsoff), "v"(keff + grow) : "memory");
      union { uint4 u; unsigned short hh[8]; } uv;
      uv.u = *reinterpret_cast<const uint4*>(vfull + grow);
#pragma unroll
      for (int i = 0; i < 8; ++i) Vt[(d + i) * 40 + s] = uv.hh[i];
    }
    asm volatile("s_wait_asynccnt 0" ::: "memory");  // K tile landed in LDS
    __syncthreads();

    if (s0 > twb + 15) continue;   // tile entirely above causal diagonal for this wave

    // scores: Q (16x192) @ K_tile^T (192x32) -> two 16x16 f32 tiles
    v8f sc0 = zero8(), sc1 = zero8();
    const unsigned short* K0 = Kl + cL * 200;
    const unsigned short* K1 = Kl + (16 + cL) * 200;
#pragma unroll
    for (int kc = 0; kc < 6; ++kc) {
      v16bf b0 = load_frag_b(K0, kc * 32, lane);
      v16bf b1 = load_frag_b(K1, kc * 32, lane);
      sc0 = WMMA_BF16(qf[kc], b0, sc0);
      sc1 = WMMA_BF16(qf[kc], b1, sc1);
    }

    // online softmax (per-row stats replicated across 16-lane group)
    float corr[8];
#pragma unroll
    for (int j = 0; j < 8; ++j) {
      int t = twb + rOff + j;
      float v0 = sc0[j] * scale; if (s0 + cL      > t) v0 = -1e30f;
      float v1 = sc1[j] * scale; if (s0 + 16 + cL > t) v1 = -1e30f;
      float mx = fmaxf(v0, v1);
      mx = fmaxf(mx, __shfl_xor(mx, 1, 32));
      mx = fmaxf(mx, __shfl_xor(mx, 2, 32));
      mx = fmaxf(mx, __shfl_xor(mx, 4, 32));
      mx = fmaxf(mx, __shfl_xor(mx, 8, 32));
      float mn = fmaxf(mrow[j], mx);
      float cr = __expf(mrow[j] - mn);
      float e0 = __expf(v0 - mn);
      float e1 = __expf(v1 - mn);
      float sm = e0 + e1;
      sm += __shfl_xor(sm, 1, 32);
      sm += __shfl_xor(sm, 2, 32);
      sm += __shfl_xor(sm, 4, 32);
      sm += __shfl_xor(sm, 8, 32);
      lrow[j] = lrow[j] * cr + sm;
      mrow[j] = mn;
      corr[j] = cr;
      Pw[(rOff + j) * 40 + cL]      = f2bf(e0);
      Pw[(rOff + j) * 40 + 16 + cL] = f2bf(e1);
    }
    asm volatile("s_wait_dscnt 0" ::: "memory");  // wave-private LDS RAW fence

#pragma unroll
    for (int ti = 0; ti < 12; ++ti)
#pragma unroll
      for (int j = 0; j < 8; ++j) ctxa[ti][j] *= corr[j];

    // ctx += P (16x32) @ V_tile (32x192)
    v16bf pa = load_frag_a(Pw + cL * 40, 0, lane);
#pragma unroll
    for (int ti = 0; ti < 12; ++ti) {
      v16bf bv = load_frag_b(Vt + (ti * 16 + cL) * 40, 0, lane);
      ctxa[ti] = WMMA_BF16(pa, bv, ctxa[ti]);
    }
  }

  // finalize: divide by l, store bf16 context
#pragma unroll
  for (int j = 0; j < 8; ++j) lrow[j] = 1.0f / lrow[j];
#pragma unroll
  for (int ti = 0; ti < 12; ++ti)
#pragma unroll
    for (int j = 0; j < 8; ++j) {
      int t = twb + rOff + j;
      size_t o = (rowB + t) * (size_t)D_MODEL + colh + ti * 16 + cL;
      ctx[o] = f2bf(ctxa[ti][j] * lrow[j]);
    }
}

/* ---------------- host launcher ---------------- */
extern "C" void kernel_launch(void* const* d_in, const int* in_sizes, int n_in,
                              void* d_out, int out_size, void* d_ws, size_t ws_size,
                              hipStream_t stream) {
  const float* x    = (const float*)d_in[0];
  const float* Wq   = (const float*)d_in[1];
  const float* Wkv  = (const float*)d_in[2];
  const float* Wuk  = (const float*)d_in[3];
  const float* Wuv  = (const float*)d_in[4];
  const float* Wout = (const float*)d_in[5];
  const float* bout = (const float*)d_in[6];
  const float* lng  = (const float*)d_in[7];
  const float* lnb  = (const float*)d_in[8];

  float* out_f   = (float*)d_out;                       // (BT, 3072)
  float* ckv_out = out_f + (size_t)BT * D_MODEL;        // (BT, 1024)

  char* w = (char*)d_ws;
  auto carve = [&](size_t bytes) -> char* {
    char* p = w; w += (bytes + 255) & ~(size_t)255; return p;
  };
  unsigned short* xb    = (unsigned short*)carve((size_t)BT * D_MODEL * 2);
  unsigned short* wqb   = (unsigned short*)carve((size_t)D_MODEL * D_MODEL * 2);
  unsigned short* wuktb = (unsigned short*)carve((size_t)DC * D_MODEL * 2);  // W_uk^T
  unsigned short* wkvb  = (unsigned short*)carve((size_t)DC * D_MODEL * 2);
  unsigned short* wuvb  = (unsigned short*)carve((size_t)D_MODEL * DC * 2);
  unsigned short* woutb = (unsigned short*)carve((size_t)D_MODEL * D_MODEL * 2);
  unsigned short* akb   = (unsigned short*)carve((size_t)D_MODEL * DC * 2);  // Wq@Wuk
  unsigned short* ckvb  = (unsigned short*)carve((size_t)BT * DC * 2);
  unsigned short* vb    = (unsigned short*)carve((size_t)BT * D_MODEL * 2);
  unsigned short* keffb = (unsigned short*)carve((size_t)BT * D_MODEL * 2);
  unsigned short* ctxb  = (unsigned short*)carve((size_t)BT * D_MODEL * 2);
  float*          cbuf  = (float*)carve((size_t)BT * DC * 4);

  // 1) conversions
  cvt_bf16<<<4096, 256, 0, stream>>>(x,    xb,    (size_t)BT * D_MODEL);
  cvt_bf16<<<4096, 256, 0, stream>>>(Wq,   wqb,   (size_t)D_MODEL * D_MODEL);
  cvt_bf16<<<2048, 256, 0, stream>>>(Wkv,  wkvb,  (size_t)DC * D_MODEL);
  cvt_bf16<<<2048, 256, 0, stream>>>(Wuv,  wuvb,  (size_t)D_MODEL * DC);
  cvt_bf16<<<4096, 256, 0, stream>>>(Wout, woutb, (size_t)D_MODEL * D_MODEL);
  cvt_bf16_t<<<2048, 256, 0, stream>>>(Wuk, wuktb, D_MODEL, DC);

  // 2) AK = Wq @ Wuk  (3072 x 1024)
  gemm_nt<true, false><<<dim3(DC / 256, D_MODEL / 64), 256, 0, stream>>>(
      wqb, wuktb, akb, nullptr, D_MODEL, DC, D_MODEL, D_MODEL, D_MODEL, DC);

  // 3) c = x @ Wkv^T  (BT x 1024, fp32)
  gemm_nt<false, false><<<dim3(DC / 256, BT / 64), 256, 0, stream>>>(
      xb, wkvb, cbuf, nullptr, BT, DC, D_MODEL, D_MODEL, D_MODEL, DC);

  // 4) LayerNorm -> c_kv (fp32 to d_out) + bf16 copy
  ln_kernel<<<BT, 256, 0, stream>>>(cbuf, lng, lnb, ckv_out, ckvb);

  // 5) v = c_kv @ Wuv^T ; k_eff = c_kv @ AK^T   (BT x 3072 each, bf16)
  gemm_nt<true, false><<<dim3(D_MODEL / 256, BT / 64), 256, 0, stream>>>(
      ckvb, wuvb, vb, nullptr, BT, D_MODEL, DC, DC, DC, D_MODEL);
  gemm_nt<true, false><<<dim3(D_MODEL / 256, BT / 64), 256, 0, stream>>>(
      ckvb, akb, keffb, nullptr, BT, D_MODEL, DC, DC, DC, D_MODEL);

  // 6) causal flash attention -> ctx (bf16)
  mla_attn<<<dim3(SEQ / 128, BATCH * NH), 256, 0, stream>>>(xb, keffb, vb, ctxb);

  // 7) out = ctx @ Wout^T + b_out  (fp32 to d_out)
  gemm_nt<false, true><<<dim3(D_MODEL / 256, BT / 64), 256, 0, stream>>>(
      ctxb, woutb, out_f, bout, BT, D_MODEL, D_MODEL, D_MODEL, D_MODEL, D_MODEL);
}
